// MultiHeadSelfAttention_39041252720808
// MI455X (gfx1250) — matrix operand order repaired and checked by reading, not compiled, not run on hardware
//
#include <hip/hip_runtime.h>

// MI455X / gfx1250, wave32. Fused mixed-precision attention:
//  - v_wmma_f32_16x16x32_f16 everywhere, fp32 accumulate
//  - fused scores+softmax+PV kernel keeps a 32x2048 f16 prob strip in LDS
//    (CDNA5 320KB/WGP LDS), so attn HBM traffic = the compulsory 512MB output
//  - oproj stages its (already-f16) A tiles with GLOBAL_LOAD_ASYNC_TO_LDS
//    (ASYNCcnt path) when the toolchain exposes the builtin.
// Workspace (f16): q16 | k16 | v16 (each B*H*S*D) | ao16 (B*S*E) = 32 MB.

typedef __attribute__((ext_vector_type(16))) _Float16 v16h;
typedef __attribute__((ext_vector_type(8)))  float    v8f;
typedef __attribute__((ext_vector_type(4)))  int      v4i;

#define E_DIM 512
#define H_NUM 8
#define D_HEAD 64
#define B_NUM 4
#define S_LEN 2048
#define BS_ROWS (B_NUM * S_LEN)   // 8192
#define BH_NUM  (B_NUM * H_NUM)   // 32

#if __has_builtin(__builtin_amdgcn_global_load_async_to_lds_b128)
#define HAVE_ASYNC_LDS 1
typedef __attribute__((address_space(1))) v4i* g_v4i_ptr;   // global AS pointer
typedef __attribute__((address_space(3))) v4i* l_v4i_ptr;   // LDS AS pointer
#endif

__device__ __forceinline__ void wait_async_zero() {
#if __has_builtin(__builtin_amdgcn_s_wait_asynccnt)
    __builtin_amdgcn_s_wait_asynccnt(0);
#elif defined(__HIP_DEVICE_COMPILE__)
    asm volatile("s_wait_asynccnt 0" ::: "memory");
#endif
}

// ---------- WMMA fragment helpers (wave32 layouts per CDNA5 ISA 7.12.2) ----------

// A matrix 16x32 f16: lane L holds row M=m0+(L&15); halves 0..7 -> K=8*(L>>4)+h,
// halves 8..15 -> K=16+8*(L>>4)+(h-8). Two 16-byte contiguous reads.
__device__ __forceinline__ v16h frag_a(const _Float16* As, int lda, int m0, int lane) {
    const int m  = m0 + (lane & 15);
    const int ko = (lane >> 4) << 3;
    const _Float16* p0 = As + (size_t)m * lda + ko;
    const _Float16* p1 = As + (size_t)m * lda + 16 + ko;
    v16h a;
#pragma unroll
    for (int i = 0; i < 8; ++i) { a[i] = p0[i]; a[i + 8] = p1[i]; }
    return a;
}

// B matrix 32x16 f16: lane L holds col N=n0+(L&15); halves 0..7 -> K=8*(L>>4)+h,
// halves 8..15 -> K=16+8*(L>>4)+(h-8).
__device__ __forceinline__ v16h frag_b(const _Float16* Bs, int ldb, int n0, int lane) {
    const int n  = n0 + (lane & 15);
    const int ko = (lane >> 4) << 3;
    v16h b;
#pragma unroll
    for (int i = 0; i < 8; ++i) {
        b[i]     = Bs[(size_t)(ko + i) * ldb + n];
        b[i + 8] = Bs[(size_t)(16 + ko + i) * ldb + n];
    }
    return b;
}

__device__ __forceinline__ v8f wmma_f16(v16h a, v16h b, v8f c) {
    return __builtin_amdgcn_wmma_f32_16x16x32_f16(false, a, false, b, (short)0, c,
                                                  false, false);
}

// ---------- Kernel 1: y = x @ W^T + bias, scaled, scattered to [B,H,S,D] f16 ----------
__global__ void __launch_bounds__(256)
qkv_proj_kernel(const float* __restrict__ X, const float* __restrict__ W,
                const float* __restrict__ bias, _Float16* __restrict__ out,
                float scale) {
    __shared__ _Float16 As[64][32];
    __shared__ _Float16 Bs[32][128];
    const int tid  = threadIdx.x;
    const int lane = tid & 31;
    const int w    = tid >> 5;
    const int wr   = w & 1;
    const int wc   = w >> 1;
    const int row0 = blockIdx.x * 64;
    const int col0 = blockIdx.y * 128;

    v8f acc[2][2] = {};
    for (int kc = 0; kc < E_DIM; kc += 32) {
        { // stage A (x tile), fp32 -> f16
            const int r = tid >> 2;
            const int c = (tid & 3) << 3;
            const float* src = X + (size_t)(row0 + r) * E_DIM + kc + c;
#pragma unroll
            for (int i = 0; i < 8; ++i) As[r][c + i] = (_Float16)src[i];
            if (kc + 32 < E_DIM) __builtin_prefetch(src + 32, 0, 0);
        }
        { // stage B transposed: Bs[k][n] = W[col0+n][kc+k]
            const int n  = tid >> 1;
            const int kh = (tid & 1) << 4;
            const float* src = W + (size_t)(col0 + n) * E_DIM + kc + kh;
#pragma unroll
            for (int i = 0; i < 16; ++i) Bs[kh + i][n] = (_Float16)src[i];
            if (kc + 32 < E_DIM) __builtin_prefetch(src + 32, 0, 0);
        }
        __syncthreads();
        v16h a0 = frag_a(&As[0][0], 32, wr * 32 + 0,  lane);
        v16h a1 = frag_a(&As[0][0], 32, wr * 32 + 16, lane);
        v16h b0 = frag_b(&Bs[0][0], 128, wc * 32 + 0,  lane);
        v16h b1 = frag_b(&Bs[0][0], 128, wc * 32 + 16, lane);
        acc[0][0] = wmma_f16(a0, b0, acc[0][0]);
        acc[0][1] = wmma_f16(a0, b1, acc[0][1]);
        acc[1][0] = wmma_f16(a1, b0, acc[1][0]);
        acc[1][1] = wmma_f16(a1, b1, acc[1][1]);
        __syncthreads();
    }
    const int colLane = lane & 15;
    const int rowOff  = (lane >> 4) * 8;
#pragma unroll
    for (int mt = 0; mt < 2; ++mt)
#pragma unroll
        for (int nt = 0; nt < 2; ++nt)
#pragma unroll
            for (int r = 0; r < 8; ++r) {
                const int m = row0 + wr * 32 + mt * 16 + rowOff + r;
                const int n = col0 + wc * 32 + nt * 16 + colLane;
                const float v = (acc[mt][nt][r] + bias[n]) * scale;
                const int b = m >> 11, s = m & (S_LEN - 1);
                const int h = n >> 6,  d = n & (D_HEAD - 1);
                out[(((size_t)(b * H_NUM + h) * S_LEN) + s) * D_HEAD + d] = (_Float16)v;
            }
}

// ---------- Kernel 2: fused scores + softmax + P@V ----------
// One block per (bh, 32 query rows). Dynamic LDS:
//   Sc  : 32 x 2048 f16  (score / prob strip)          131072 B
//   Qt  : 64 x 32  f16  (Q^T tile for B-fragments)       4096 B
//   SAcc: 32 x 64  f32  (cross-wave PV reduction)        8192 B
#define FUSED_SMEM_BYTES (32 * S_LEN * 2 + D_HEAD * 32 * 2 + 32 * D_HEAD * 4)

__global__ void __launch_bounds__(256)
fused_attn_kernel(const _Float16* __restrict__ Q, const _Float16* __restrict__ K,
                  const _Float16* __restrict__ V, float* __restrict__ attn,
                  _Float16* __restrict__ AO) {
    extern __shared__ char smem[];
    _Float16* Sc   = (_Float16*)smem;                              // [32][2048]
    _Float16* Qt   = (_Float16*)(smem + 32 * S_LEN * 2);           // [64][32]
    float*    SAcc = (float*)(smem + 32 * S_LEN * 2 + D_HEAD * 32 * 2); // [32][64]

    const int bh = blockIdx.y;
    const int b  = bh >> 3, h = bh & 7;
    const int s0 = blockIdx.x * 32;
    const int tid  = threadIdx.x;
    const int lane = tid & 31;
    const int w    = tid >> 5;                 // 0..7
    const int colLane = lane & 15;
    const int rowOff  = (lane >> 4) * 8;

    const _Float16* Qb = Q + (size_t)bh * S_LEN * D_HEAD;
    const _Float16* Kb = K + (size_t)bh * S_LEN * D_HEAD;
    const _Float16* Vb = V + (size_t)bh * S_LEN * D_HEAD;

    // ---- stage Q^T (Qt[d][s]) and zero SAcc ----
    {
        const int s = tid >> 3, dc = (tid & 7) << 3;
        const _Float16* src = Qb + (size_t)(s0 + s) * D_HEAD + dc;
#pragma unroll
        for (int i = 0; i < 8; ++i) Qt[(dc + i) * 32 + s] = src[i];
#pragma unroll
        for (int i = 0; i < 8; ++i) SAcc[tid + i * 256] = 0.0f;
    }
    __syncthreads();

    // ---- Phase 1: scores^T = K @ Q^T  (M = t keys, N = s queries, K-dim = d) ----
    // Each wave owns a 256-wide strip of t; results stored transposed into Sc[s][t].
    {
        v16h bq[2][2];   // [kc][s-tile]
        bq[0][0] = frag_b(Qt,            32, 0,  lane);
        bq[0][1] = frag_b(Qt,            32, 16, lane);
        bq[1][0] = frag_b(Qt + 32 * 32,  32, 0,  lane);
        bq[1][1] = frag_b(Qt + 32 * 32,  32, 16, lane);
        const int tstrip = w * 256;
#pragma unroll 1
        for (int g = 0; g < 8; ++g) {
            const int t0 = tstrip + g * 32;
            v8f acc[2][2] = {};
#pragma unroll
            for (int kc2 = 0; kc2 < 2; ++kc2) {
                const _Float16* ka = Kb + (size_t)t0 * D_HEAD + kc2 * 32;
                v16h a0 = frag_a(ka, D_HEAD, 0,  lane);
                v16h a1 = frag_a(ka, D_HEAD, 16, lane);
                acc[0][0] = wmma_f16(a0, bq[kc2][0], acc[0][0]);
                acc[0][1] = wmma_f16(a0, bq[kc2][1], acc[0][1]);
                acc[1][0] = wmma_f16(a1, bq[kc2][0], acc[1][0]);
                acc[1][1] = wmma_f16(a1, bq[kc2][1], acc[1][1]);
            }
            // transposed store: C row = t (8 consecutive), C col = s -> Sc[s][t]
#pragma unroll
            for (int mt = 0; mt < 2; ++mt)
#pragma unroll
                for (int nt = 0; nt < 2; ++nt) {
                    const int s  = nt * 16 + colLane;
                    const int tt = t0 + mt * 16 + rowOff;
                    union { _Float16 hh[8]; uint4 u; } pk;
#pragma unroll
                    for (int r = 0; r < 8; ++r) pk.hh[r] = (_Float16)acc[mt][nt][r];
                    *(uint4*)(&Sc[(size_t)s * S_LEN + tt]) = pk.u;
                }
        }
    }
    __syncthreads();

    // ---- Phase 2: softmax per row; write fp32 probs to d_out, f16 back to Sc ----
    {
#pragma unroll 1
        for (int rr = 0; rr < 4; ++rr) {
            const int srow = w * 4 + rr;
            _Float16* rowp = Sc + (size_t)srow * S_LEN + lane * 64;
            float vals[64];
            float m = -3.4e38f;
#pragma unroll
            for (int i = 0; i < 64; ++i) { vals[i] = (float)rowp[i]; m = fmaxf(m, vals[i]); }
#pragma unroll
            for (int off = 16; off > 0; off >>= 1) m = fmaxf(m, __shfl_xor(m, off));
            float sum = 0.0f;
#pragma unroll
            for (int i = 0; i < 64; ++i) { vals[i] = __expf(vals[i] - m); sum += vals[i]; }
#pragma unroll
            for (int off = 16; off > 0; off >>= 1) sum += __shfl_xor(sum, off);
            const float inv = 1.0f / sum;
            float* gp = attn + ((size_t)bh * S_LEN + s0 + srow) * S_LEN + lane * 64;
#pragma unroll
            for (int i = 0; i < 64; ++i) {
                const float p = vals[i] * inv;
                gp[i]   = p;
                rowp[i] = (_Float16)p;
            }
        }
    }
    __syncthreads();

    // ---- Phase 3: out = P @ V, K-split across 8 waves, ds_add_f32 reduction ----
    {
        const int tbase = w * 256;
        v8f acc[2][4] = {};
#pragma unroll 1
        for (int kk = 0; kk < 8; ++kk) {
            const int tc = tbase + kk * 32;
            const _Float16* pA = Sc + tc;                 // A[s][t], lda = S_LEN
            v16h a0 = frag_a(pA, S_LEN, 0,  lane);
            v16h a1 = frag_a(pA, S_LEN, 16, lane);
            const _Float16* pB = Vb + (size_t)tc * D_HEAD; // B rows = t, cols = d
#pragma unroll
            for (int nt = 0; nt < 4; ++nt) {
                v16h bv = frag_b(pB, D_HEAD, nt * 16, lane);
                acc[0][nt] = wmma_f16(a0, bv, acc[0][nt]);
                acc[1][nt] = wmma_f16(a1, bv, acc[1][nt]);
            }
        }
#pragma unroll
        for (int mt = 0; mt < 2; ++mt)
#pragma unroll
            for (int nt = 0; nt < 4; ++nt)
#pragma unroll
                for (int r = 0; r < 8; ++r) {
                    const int s = mt * 16 + rowOff + r;
                    const int d = nt * 16 + colLane;
                    atomicAdd(&SAcc[s * D_HEAD + d], acc[mt][nt][r]);
                }
    }
    __syncthreads();

    // ---- writeback attn_out tile into [B,S,E] f16 ----
    {
#pragma unroll
        for (int i = 0; i < 8; ++i) {
            const int idx = tid * 8 + i;
            const int s = idx >> 6, d = idx & 63;
            AO[((size_t)b * S_LEN + s0 + s) * E_DIM + h * D_HEAD + d] =
                (_Float16)SAcc[s * D_HEAD + d];
        }
    }
}

// ---------- Kernel 3: out = ao @ Wo^T + bo, fp32 to d_out[0 .. B*S*E) ----------
__global__ void __launch_bounds__(256)
oproj_kernel(const _Float16* __restrict__ A, const float* __restrict__ W,
             const float* __restrict__ bias, float* __restrict__ Out) {
    __shared__ _Float16 As[64][32];
    __shared__ _Float16 Bs[32][128];
    const int tid  = threadIdx.x;
    const int lane = tid & 31;
    const int w    = tid >> 5;
    const int wr   = w & 1;
    const int wc   = w >> 1;
    const int row0 = blockIdx.x * 64;
    const int col0 = blockIdx.y * 128;

    v8f acc[2][2] = {};
    for (int kc = 0; kc < E_DIM; kc += 32) {
        // stage A (f16 tile): async global->LDS when available (ASYNCcnt path)
        {
            const int r = tid >> 2, c = (tid & 3) << 3;
            const _Float16* src = A + (size_t)(row0 + r) * E_DIM + kc + c;
#ifdef HAVE_ASYNC_LDS
            __builtin_amdgcn_global_load_async_to_lds_b128(
                (g_v4i_ptr)src, (l_v4i_ptr)&As[r][c], 0, 0);
#else
#pragma unroll
            for (int i = 0; i < 8; ++i) As[r][c + i] = src[i];
#endif
        }
        { // stage B transposed with fp32 -> f16 conversion (needs VGPR staging)
            const int n = tid >> 1, kh = (tid & 1) << 4;
            const float* src = W + (size_t)(col0 + n) * E_DIM + kc + kh;
#pragma unroll
            for (int i = 0; i < 16; ++i) Bs[kh + i][n] = (_Float16)src[i];
            if (kc + 32 < E_DIM) __builtin_prefetch(src + 32, 0, 0);
        }
#ifdef HAVE_ASYNC_LDS
        wait_async_zero();
#endif
        __syncthreads();
        v16h a0 = frag_a(&As[0][0], 32, wr * 32 + 0,  lane);
        v16h a1 = frag_a(&As[0][0], 32, wr * 32 + 16, lane);
        v16h b0 = frag_b(&Bs[0][0], 128, wc * 32 + 0,  lane);
        v16h b1 = frag_b(&Bs[0][0], 128, wc * 32 + 16, lane);
        acc[0][0] = wmma_f16(a0, b0, acc[0][0]);
        acc[0][1] = wmma_f16(a0, b1, acc[0][1]);
        acc[1][0] = wmma_f16(a1, b0, acc[1][0]);
        acc[1][1] = wmma_f16(a1, b1, acc[1][1]);
        __syncthreads();
    }
    const int colLane = lane & 15;
    const int rowOff  = (lane >> 4) * 8;
#pragma unroll
    for (int mt = 0; mt < 2; ++mt)
#pragma unroll
        for (int nt = 0; nt < 2; ++nt)
#pragma unroll
            for (int r = 0; r < 8; ++r) {
                const int m = row0 + wr * 32 + mt * 16 + rowOff + r;
                const int n = col0 + wc * 32 + nt * 16 + colLane;
                Out[(size_t)m * E_DIM + n] = acc[mt][nt][r] + bias[n];
            }
}

// ---------- Host launch ----------
extern "C" void kernel_launch(void* const* d_in, const int* in_sizes, int n_in,
                              void* d_out, int out_size, void* d_ws, size_t ws_size,
                              hipStream_t stream) {
    (void)in_sizes; (void)n_in; (void)out_size; (void)ws_size;
    const float* x  = (const float*)d_in[0];
    const float* Wq = (const float*)d_in[1];
    const float* bq = (const float*)d_in[2];
    const float* Wk = (const float*)d_in[3];
    const float* bk = (const float*)d_in[4];
    const float* Wv = (const float*)d_in[5];
    const float* bv = (const float*)d_in[6];
    const float* Wo = (const float*)d_in[7];
    const float* bo = (const float*)d_in[8];

    float* out  = (float*)d_out;                        // [B,S,E]
    float* attn = out + (size_t)B_NUM * S_LEN * E_DIM;  // [B,H,S,S]

    const size_t qkvElems = (size_t)BH_NUM * S_LEN * D_HEAD;
    _Float16* q16  = (_Float16*)d_ws;
    _Float16* k16  = q16 + qkvElems;
    _Float16* v16  = k16 + qkvElems;
    _Float16* ao16 = v16 + qkvElems;                    // [B,S,E]

    const dim3 blk(256);
    const dim3 gProj(BS_ROWS / 64, E_DIM / 128);        // 128 x 4
    const float qscale = 0.125f;                         // 1/sqrt(64) folded into q

    qkv_proj_kernel<<<gProj, blk, 0, stream>>>(x, Wq, bq, q16, qscale);
    qkv_proj_kernel<<<gProj, blk, 0, stream>>>(x, Wk, bk, k16, 1.0f);
    qkv_proj_kernel<<<gProj, blk, 0, stream>>>(x, Wv, bv, v16, 1.0f);

    fused_attn_kernel<<<dim3(S_LEN / 32, BH_NUM), blk, FUSED_SMEM_BYTES, stream>>>(
        q16, k16, v16, attn, ao16);

    oproj_kernel<<<gProj, blk, 0, stream>>>(ao16, Wo, bo, out);
}
